// MLPFactorGraphLayer_49306224558820
// MI455X (gfx1250) — matrix operand order repaired and checked by reading, not compiled, not run on hardware
//
#include <hip/hip_runtime.h>
#include <hip/hip_bf16.h>

// Problem sizes (fixed by the reference)
#define E_N 500000
#define V_N 50000
#define F_N 100000
#define D_N 128
// 3*D = 384 concat features -> 12 K-steps of 32 (bf16 WMMA K)
#define KSTEPS 12
#define NTILES 8                                   // 128 outputs / 16
#define WFRAG_ELEMS (NTILES * KSTEPS * 32 * 16)    // packed bf16 weight elements
#define WFRAG_BYTES (WFRAG_ELEMS * 2)              // 98304 B = 96 KB (fits in WGP LDS)
#define MLP_BLOCKS 1024                            // persistent grid for weight reuse

typedef __attribute__((ext_vector_type(16))) __bf16 bf16x16;
typedef __attribute__((ext_vector_type(8)))  float  f32x8;
typedef __attribute__((ext_vector_type(4)))  float  f32x4;
typedef __attribute__((ext_vector_type(4)))  unsigned int u32x4;

__device__ __forceinline__ void load8(const float* __restrict__ p, float* v) {
    f32x4 a = *(const f32x4*)(p);
    f32x4 b = *(const f32x4*)(p + 4);
#pragma unroll
    for (int i = 0; i < 4; ++i) { v[i] = a[i]; v[4 + i] = b[i]; }
}

// ---------------------------------------------------------------------------
// Utility kernels
// ---------------------------------------------------------------------------
__global__ __launch_bounds__(256) void zero_f32(float* __restrict__ p, long n4) {
    long i = (long)blockIdx.x * blockDim.x + threadIdx.x;
    long stride = (long)gridDim.x * blockDim.x;
    f32x4 z = {0.f, 0.f, 0.f, 0.f};
    for (; i < n4; i += stride) ((f32x4*)p)[i] = z;
}

__global__ __launch_bounds__(256) void add_rows(const float* __restrict__ a,
                                                const float* __restrict__ b,
                                                float* __restrict__ o, long n4) {
    long i = (long)blockIdx.x * blockDim.x + threadIdx.x;
    long stride = (long)gridDim.x * blockDim.x;
    for (; i < n4; i += stride) {
        f32x4 x = ((const f32x4*)a)[i];
        f32x4 y = ((const f32x4*)b)[i];
        ((f32x4*)o)[i] = x + y;
    }
}

// segment_sum: sum[idx[e]] += msg[e]  (per 4-float chunk, f32 atomics hit L2)
__global__ __launch_bounds__(256) void scatter_add(const float* __restrict__ msg,
                                                   const int* __restrict__ idx,
                                                   float* __restrict__ sum) {
    long t = (long)blockIdx.x * blockDim.x + threadIdx.x;
    const long total = (long)E_N * (D_N / 4);
    if (t >= total) return;
    int e = (int)(t >> 5);          // D/4 = 32 chunks per row
    int c = (int)(t & 31) << 2;
    int v = idx[e];
    f32x4 m = *(const f32x4*)(msg + (size_t)e * D_N + c);
    float* dst = sum + (size_t)v * D_N + c;
    atomicAdd(dst + 0, m[0]);
    atomicAdd(dst + 1, m[1]);
    atomicAdd(dst + 2, m[2]);
    atomicAdd(dst + 3, m[3]);
}

// Pack W [384,128] f32 (row-major, k-major) into bf16 WMMA B-fragments.
// Fragment (nt,ks) lane L holds N = nt*16 + (L&15), K = ks*32 + (i>>3)*16 + (L>>4)*8 + (i&7).
__global__ __launch_bounds__(256) void pack_w(const float* __restrict__ W,
                                              __bf16* __restrict__ out) {
    int t = blockIdx.x * blockDim.x + threadIdx.x;
    if (t >= NTILES * KSTEPS * 32) return;
    int lane = t & 31;
    int rest = t >> 5;
    int ks = rest % KSTEPS;
    int nt = rest / KSTEPS;
    int n = nt * 16 + (lane & 15);
    int khalf = lane >> 4;
    __bf16* dst = out + (size_t)t * 16;
#pragma unroll
    for (int i = 0; i < 16; ++i) {
        int k = ks * 32 + (i >> 3) * 16 + khalf * 8 + (i & 7);
        dst[i] = (__bf16)W[(size_t)k * D_N + n];
    }
}

// ---------------------------------------------------------------------------
// Fused edge MLP: h = relu([node[idx[e]], sum[idx[e]]-sub[e], ea[e]] @ W + b)
// Packed bf16 weights (96 KB) are staged once per block into LDS, then each
// wave grid-strides over 16-edge x 128-out tiles, feeding B fragments from LDS
// (ds_load_b128) into v_wmma_f32_16x16x32_bf16. Optionally scatter-adds h.
// ---------------------------------------------------------------------------
template <bool ACCUM>
__global__ __launch_bounds__(256) void mlp_edge(
    const float* __restrict__ nodeSrc,      // [*,D] gathered via gatherIdx
    const int* __restrict__ gatherIdx,      // [E]
    const float* __restrict__ sumBuf,       // [*,D] gathered via gatherIdx
    const float* __restrict__ subBuf,       // [E,D]
    const float* __restrict__ edgeAttr,     // [E,D]
    const u32x4* __restrict__ pW,           // packed bf16 weights (global)
    const float* __restrict__ bias,         // [D]
    float* __restrict__ outBuf,             // [E,D]
    float* __restrict__ accumBuf,           // [*,D] (ACCUM only)
    const int* __restrict__ accumIdx,       // [E]   (ACCUM only)
    int numTiles) {
    extern __shared__ char smem[];
    u32x4* ldsW4 = (u32x4*)smem;
    const bf16x16* ldsW = (const bf16x16*)smem;

    // Stage packed weights global -> LDS (cooperative, 16B per transfer)
    {
        const int nchunks = WFRAG_BYTES / 16;   // 6144
        for (int i = threadIdx.x; i < nchunks; i += 256) ldsW4[i] = pW[i];
    }
    __syncthreads();

    const int waveInBlk = threadIdx.x >> 5;
    const int lane = threadIdx.x & 31;
    const int row = lane & 15;              // edge-in-tile this lane feeds for A
    const int khalf = lane >> 4;
    const int waveStride = gridDim.x * 8;

    for (int tile = blockIdx.x * 8 + waveInBlk; tile < numTiles; tile += waveStride) {
        const int e0 = tile * 16;
        const int e = e0 + row;

        const int gi = gatherIdx[e];
        const float* nodeRow = nodeSrc + (size_t)gi * D_N;
        const float* sumRow  = sumBuf  + (size_t)gi * D_N;
        const float* subRow  = subBuf  + (size_t)e  * D_N;
        const float* eaRow   = edgeAttr + (size_t)e * D_N;

        f32x8 acc[NTILES];
        f32x8 zero = {0.f, 0.f, 0.f, 0.f, 0.f, 0.f, 0.f, 0.f};
#pragma unroll
        for (int nt = 0; nt < NTILES; ++nt) acc[nt] = zero;

#pragma unroll
        for (int ks = 0; ks < KSTEPS; ++ks) {
            // Each K=32 step lies fully inside one concat region (384 = 3*128),
            // so this branch is wave-uniform (EXEC stays all-1s for WMMA).
            const int region = ks >> 2;                       // 0:node 1:sum-sub 2:ea
            const int off0 = ((ks & 3) << 5) + (khalf << 3);  // start of run 0
            float av[16];
            if (region == 0) {
                load8(nodeRow + off0, av);
                load8(nodeRow + off0 + 16, av + 8);
            } else if (region == 1) {
                float s[8], m[8];
                load8(sumRow + off0, s);
                load8(subRow + off0, m);
#pragma unroll
                for (int i = 0; i < 8; ++i) av[i] = s[i] - m[i];
                load8(sumRow + off0 + 16, s);
                load8(subRow + off0 + 16, m);
#pragma unroll
                for (int i = 0; i < 8; ++i) av[8 + i] = s[i] - m[i];
            } else {
                load8(eaRow + off0, av);
                load8(eaRow + off0 + 16, av + 8);
            }
            bf16x16 afrag;
#pragma unroll
            for (int i = 0; i < 16; ++i) afrag[i] = (__bf16)av[i];

#pragma unroll
            for (int nt = 0; nt < NTILES; ++nt) {
                bf16x16 bfrag = ldsW[((size_t)nt * KSTEPS + ks) * 32 + lane];
                acc[nt] = __builtin_amdgcn_wmma_f32_16x16x32_bf16(
                    false, afrag, false, bfrag, (short)0, acc[nt], false, false);
            }
        }

        int fidx[8];
        if constexpr (ACCUM) {
#pragma unroll
            for (int r = 0; r < 8; ++r) fidx[r] = accumIdx[e0 + r + (khalf << 3)];
        }

        // C layout: lane L -> N = L&15 (+ntile*16); VGPR r -> M = r + 8*(L>>4)
#pragma unroll
        for (int nt = 0; nt < NTILES; ++nt) {
            const int n = nt * 16 + (lane & 15);
            const float bn = bias[n];
#pragma unroll
            for (int r = 0; r < 8; ++r) {
                const int er = e0 + r + (khalf << 3);
                float v = acc[nt][r] + bn;
                v = v > 0.f ? v : 0.f;
                outBuf[(size_t)er * D_N + n] = v;
                if constexpr (ACCUM) {
                    atomicAdd(accumBuf + (size_t)fidx[r] * D_N + n, v);
                }
            }
        }
    }
}

// ---------------------------------------------------------------------------
// Host launcher
// ---------------------------------------------------------------------------
extern "C" void kernel_launch(void* const* d_in, const int* in_sizes, int n_in,
                              void* d_out, int out_size, void* d_ws, size_t ws_size,
                              hipStream_t stream) {
    const float* prev_m    = (const float*)d_in[0];  // [E,D]
    const float* variable  = (const float*)d_in[1];  // [V,D]
    const float* factor    = (const float*)d_in[2];  // [F,D]
    const float* edge_attr = (const float*)d_in[3];  // [E,D]
    const int*   v_to_f    = (const int*)d_in[4];    // [E] -> V
    const int*   f_to_v    = (const int*)d_in[5];    // [E] -> F
    const float* Wv        = (const float*)d_in[6];  // [3D,D]
    const float* bv        = (const float*)d_in[7];  // [D]
    const float* Wf        = (const float*)d_in[8];  // [3D,D]
    const float* bf        = (const float*)d_in[9];  // [D]

    float* out_m        = (float*)d_out;                    // [E,D]
    float* out_factor   = out_m + (size_t)E_N * D_N;        // [F,D]
    float* out_variable = out_factor + (size_t)F_N * D_N;   // [V,D]

    // Workspace layout
    float* sum_f_to_v = (float*)d_ws;                               // [V,D]
    float* sum_v_to_f = sum_f_to_v + (size_t)V_N * D_N;             // [F,D] (contiguous)
    float* m_v_to_f   = sum_v_to_f + (size_t)F_N * D_N;             // [E,D]
    __bf16* pWv = (__bf16*)(m_v_to_f + (size_t)E_N * D_N);          // packed weights
    __bf16* pWf = pWv + WFRAG_ELEMS;

    // 1) zero the segment accumulators (one pass, they are contiguous)
    long zn4 = ((long)(V_N + F_N) * D_N) / 4;
    zero_f32<<<2048, 256, 0, stream>>>(sum_f_to_v, zn4);

    // 2) pack weights into bf16 WMMA B-fragment order
    pack_w<<<(NTILES * KSTEPS * 32 + 255) / 256, 256, 0, stream>>>(Wv, pWv);
    pack_w<<<(NTILES * KSTEPS * 32 + 255) / 256, 256, 0, stream>>>(Wf, pWf);

    // 3) sum_f_to_v = segment_sum(prev_m, v_to_f)
    long st = (long)E_N * (D_N / 4);
    scatter_add<<<(int)((st + 255) / 256), 256, 0, stream>>>(prev_m, v_to_f, sum_f_to_v);

    // 4) m_v_to_f = relu([variable[v], sum_f_to_v[v]-prev_m, ea] @ Wv + bv);
    //    sum_v_to_f += m_v_to_f (scattered by f_to_v)
    const int tiles = E_N / 16;  // 31250 exactly
    mlp_edge<true><<<MLP_BLOCKS, 256, WFRAG_BYTES, stream>>>(
        variable, v_to_f, sum_f_to_v, prev_m, edge_attr,
        (const u32x4*)pWv, bv, m_v_to_f, sum_v_to_f, f_to_v, tiles);

    // 5) residual outputs
    add_rows<<<2048, 256, 0, stream>>>(sum_f_to_v, variable, out_variable,
                                       (long)V_N * D_N / 4);
    add_rows<<<2048, 256, 0, stream>>>(sum_v_to_f, factor, out_factor,
                                       (long)F_N * D_N / 4);

    // 6) m_f_to_v = relu([factor[f], sum_v_to_f[f]-m_v_to_f, ea] @ Wf + bf)
    mlp_edge<false><<<MLP_BLOCKS, 256, WFRAG_BYTES, stream>>>(
        factor, f_to_v, sum_v_to_f, m_v_to_f, edge_attr,
        (const u32x4*)pWf, bf, out_m, nullptr, nullptr, tiles);
}
